// GraphormerAttention_45844480917697
// MI455X (gfx1250) — matrix operand order repaired and checked by reading, not compiled, not run on hardware
//
#include <hip/hip_runtime.h>
#include <hip/hip_bf16.h>

// ---------------------------------------------------------------------------
// Graphormer attention forward, gfx1250 (CDNA5), wave32 + WMMA f16.
// B=16, N=1024, D=512, H=8, HD=64.
// ---------------------------------------------------------------------------

typedef __attribute__((ext_vector_type(16))) _Float16 v16h;
typedef __attribute__((ext_vector_type(8)))  _Float16 v8h;
typedef __attribute__((ext_vector_type(4)))  _Float16 v4h;
typedef __attribute__((ext_vector_type(8)))  float    v8f;

#define BATCH 16
#define NSEQ  1024
#define DIM   512
#define NHEAD 8
#define HDIM  64

__device__ __forceinline__ v8f wmma_f16(v16h a, v16h b, v8f c) {
  // D = A(16x32,f16) * B(32x16,f16) + C(16x16,f32)
  return __builtin_amdgcn_wmma_f32_16x16x32_f16(
      /*neg_a=*/false, a, /*neg_b=*/false, b,
      /*c_mod=*/(short)0, c, /*reuse_a=*/false, /*reuse_b=*/false);
}

// A-operand (16x32 f16 tile, rows m0..m0+15, k columns k0..k0+31) gathered
// from a row-major f16 matrix. lane<16 -> K {0..15},
// lane>=16 -> K {16..31}; loaded as two contiguous 8-half (16B) runs.
__device__ __forceinline__ v16h load_a_f16(const _Float16* base, int stride,
                                           int m0, int k0, int lane) {
  int row = m0 + (lane & 15);
  int b8  = (lane >= 16) ? 16 : 0;
  const _Float16* p = base + (size_t)row * stride + k0 + b8;
  v8h lo = *(const v8h*)p;
  v8h hi = *(const v8h*)(p + 8);
  v16h r;
  #pragma unroll
  for (int i = 0; i < 8; ++i) { r[i] = lo[i]; r[i + 8] = hi[i]; }
  return r;
}

// B-operand (32x16 f16 tile) where the source matrix is stored K-major,
// i.e. src[n][k] row-major with row stride `stride`. Lane holds column
// n0+(lane&15); K values are contiguous (+16 for upper half-wave):
// one aligned 32-byte vector load.
__device__ __forceinline__ v16h load_b_tr_f16(const _Float16* base, int stride,
                                              int n0, int k0, int lane) {
  int col = n0 + (lane & 15);
  int kb  = k0 + ((lane >= 16) ? 16 : 0);
  return *(const v16h*)(base + (size_t)col * stride + kb);
}

// ---------------------------------------------------------------------------
// Kernel 0: f32 -> f16 streaming conversion (x, w_qkv, w_proj staging).
// n must be a multiple of 4 (all our sizes are multiples of 1024).
// ---------------------------------------------------------------------------
__global__ __launch_bounds__(256) void cvt_f16_kernel(
    const float* __restrict__ in, _Float16* __restrict__ out, int n) {
  int i = (blockIdx.x * 256 + threadIdx.x) * 4;
  if (i < n) {
    float4 v = *(const float4*)(in + i);
    v4h o;
    o[0] = (_Float16)v.x; o[1] = (_Float16)v.y;
    o[2] = (_Float16)v.z; o[3] = (_Float16)v.w;
    *(v4h*)(out + i) = o;
  }
}

// ---------------------------------------------------------------------------
// Kernel 1: QKV projection.  y[m,n] = sum_k xh[m,k] * wh[n,k]   (f16 WMMA)
// M=16384, K=512, Nout=1536. Workgroup = 64x128 output tile, 8 waves,
// each wave a 32x32 sub-tile (4 WMMA accumulators, shared A/B loads).
// Epilogue scatters f16 to Q [b,h,n,hd], K [b,h,n,hd], Vt [b,h,hd,n].
// ---------------------------------------------------------------------------
__global__ __launch_bounds__(256) void qkv_kernel(
    const _Float16* __restrict__ xh, const _Float16* __restrict__ wh,
    _Float16* __restrict__ Qh, _Float16* __restrict__ Kh,
    _Float16* __restrict__ Vt) {
  int lane = threadIdx.x & 31;
  int w    = threadIdx.x >> 5;
  int m0 = blockIdx.x * 64  + (w & 1) * 32;
  int n0 = blockIdx.y * 128 + (w >> 1) * 32;

  v8f acc00 = {}, acc01 = {}, acc10 = {}, acc11 = {};
  #pragma unroll 4
  for (int k0 = 0; k0 < DIM; k0 += 32) {
    v16h a0 = load_a_f16(xh, DIM, m0,      k0, lane);
    v16h a1 = load_a_f16(xh, DIM, m0 + 16, k0, lane);
    v16h b0 = load_b_tr_f16(wh, DIM, n0,      k0, lane);
    v16h b1 = load_b_tr_f16(wh, DIM, n0 + 16, k0, lane);
    acc00 = wmma_f16(a0, b0, acc00);
    acc01 = wmma_f16(a0, b1, acc01);
    acc10 = wmma_f16(a1, b0, acc10);
    acc11 = wmma_f16(a1, b1, acc11);
  }

  #pragma unroll
  for (int tm = 0; tm < 2; ++tm) {
    #pragma unroll
    for (int tn = 0; tn < 2; ++tn) {
      v8f acc = tm ? (tn ? acc11 : acc10) : (tn ? acc01 : acc00);
      int n  = n0 + tn * 16 + (lane & 15);
      int s  = n >> 9;          // 0=q, 1=k, 2=v
      int h  = (n >> 6) & 7;
      int hd = n & 63;
      int mbase = m0 + tm * 16 + ((lane >= 16) ? 8 : 0);
      #pragma unroll
      for (int r = 0; r < 8; ++r) {
        int m = mbase + r;                 // global token = b*N + q
        int b = m >> 10, q = m & 1023;
        _Float16 val = (_Float16)acc[r];
        size_t bh = (size_t)(b * NHEAD + h);
        if (s == 0)      Qh[(bh * NSEQ + q) * HDIM + hd] = val;
        else if (s == 1) Kh[(bh * NSEQ + q) * HDIM + hd] = val;
        else             Vt[(bh * HDIM + hd) * NSEQ + q] = val;
      }
    }
  }
}

// ---------------------------------------------------------------------------
// Kernel 2: fused attention for one (b,h,16-query tile).
// LDS: S[16][1024] f32 (64KB) + Obuf[16][64] f32 (4KB) + P[16][1024] f16
// (32KB), dynamic (100KB total).
// Phase 1: S = Q Kt * scale (WMMA).   Phase 2: + bias, mask (coalesced).
// Phase 3: softmax per row (wave shfl reductions + v_exp), write
//          attn_weights (coalesced f32 stream) and P (f16, LDS).
// Phase 4: O = P V (WMMA, A = ds_load_b128 from P, B = vec load from Vt),
//          ds_add_f32 cross-wave K reduction.
// Phase 5: store O as f16 [b][q][h][hd].
// ---------------------------------------------------------------------------
__global__ __launch_bounds__(256) void attn_kernel(
    const float* __restrict__ attn_bias, const unsigned char* __restrict__ mask,
    const _Float16* __restrict__ Qh, const _Float16* __restrict__ Kh,
    const _Float16* __restrict__ Vt, float* __restrict__ attn_out,
    _Float16* __restrict__ Oh) {
  extern __shared__ char smem[];
  float*    S    = (float*)smem;                                   // 16*1024 f32
  float*    Obuf = (float*)(smem + 16 * NSEQ * sizeof(float));     // 16*64 f32
  _Float16* P    = (_Float16*)(smem + (16 * NSEQ + 16 * HDIM) * sizeof(float));

  int tid = threadIdx.x, lane = tid & 31, w = tid >> 5;
  int bid = blockIdx.x;
  int qt = bid & 63;          // query tile (N/16 = 64)
  int bh = bid >> 6;          // b*H + h
  int b  = bh >> 3;
  int h  = bh & 7;
  int q0 = qt * 16;

  const _Float16* Qb = Qh + (size_t)bh * NSEQ * HDIM;
  const _Float16* Kb = Kh + (size_t)bh * NSEQ * HDIM;
  const _Float16* Vb = Vt + (size_t)bh * HDIM * NSEQ;

  // ---- Phase 1: scores. Wave w covers key tiles w*8 .. w*8+7.
  v16h aq0 = load_a_f16(Qb, HDIM, q0, 0,  lane);
  v16h aq1 = load_a_f16(Qb, HDIM, q0, 32, lane);
  int mb = (lane >= 16) ? 8 : 0;
  #pragma unroll 2
  for (int t = 0; t < 8; ++t) {
    int kt0 = (w * 8 + t) * 16;
    v16h bk0 = load_b_tr_f16(Kb, HDIM, kt0, 0,  lane);
    v16h bk1 = load_b_tr_f16(Kb, HDIM, kt0, 32, lane);
    v8f acc = {};
    acc = wmma_f16(aq0, bk0, acc);
    acc = wmma_f16(aq1, bk1, acc);
    int col = kt0 + (lane & 15);
    #pragma unroll
    for (int r = 0; r < 8; ++r)
      S[(mb + r) * NSEQ + col] = acc[r] * 0.125f;   // HD^-0.5 = 1/8
  }
  __syncthreads();

  // ---- Phase 2: bias + mask (fully coalesced streaming of attn_bias).
  const float* biasb = attn_bias + ((size_t)b * NSEQ + q0) * NSEQ;
  const unsigned char* maskb = mask + (size_t)b * NSEQ;
  for (int i = tid; i < 16 * NSEQ; i += 256) {
    int row = i >> 10, col = i & 1023;
    float v = S[i] + biasb[(size_t)row * NSEQ + col];
    if (!maskb[col]) v = -__builtin_inff();
    S[i] = v;
  }
  __syncthreads();

  // ---- Phase 3: softmax, wave w handles rows 2w, 2w+1.
  float* attn_b = attn_out + ((size_t)bh * NSEQ + q0) * NSEQ;
  #pragma unroll
  for (int rr = 0; rr < 2; ++rr) {
    int row = w * 2 + rr;
    float mx = -__builtin_inff();
    for (int c = lane; c < NSEQ; c += 32) mx = fmaxf(mx, S[row * NSEQ + c]);
    #pragma unroll
    for (int o = 16; o > 0; o >>= 1) mx = fmaxf(mx, __shfl_xor(mx, o, 32));
    float sum = 0.f;
    for (int c = lane; c < NSEQ; c += 32) {
      float e = __expf(S[row * NSEQ + c] - mx);
      S[row * NSEQ + c] = e;
      sum += e;
    }
    #pragma unroll
    for (int o = 16; o > 0; o >>= 1) sum += __shfl_xor(sum, o, 32);
    float inv = 1.0f / sum;
    for (int c = lane; c < NSEQ; c += 32) {
      float p = S[row * NSEQ + c] * inv;
      P[row * NSEQ + c] = (_Float16)p;            // f16 copy for the PV GEMM
      attn_b[(size_t)row * NSEQ + c] = p;         // coalesced f32 stream out
    }
  }
  for (int i = tid; i < 16 * HDIM; i += 256) Obuf[i] = 0.f;
  __syncthreads();

  // ---- Phase 4: O = P V. Wave w: hd tile = w&3, K half = w>>2.
  int hd0 = (w & 3) * 16;
  int ks  = (w >> 2) * 512;
  v8f acc = {};
  for (int k0 = 0; k0 < 512; k0 += 32) {
    int kb = ks + k0;
    v16h a  = load_a_f16(P, NSEQ, 0, kb, lane);     // ds_load_b128 pair
    v16h bv = load_b_tr_f16(Vb, NSEQ, hd0, kb, lane);
    acc = wmma_f16(a, bv, acc);
  }
  {
    int col = hd0 + (lane & 15);
    #pragma unroll
    for (int r = 0; r < 8; ++r)
      atomicAdd(&Obuf[(mb + r) * HDIM + col], acc[r]);  // ds_add_f32
  }
  __syncthreads();

  // ---- Phase 5: O -> f16 staging [b][q][h][hd] (== [m][512] row-major).
  for (int i = tid; i < 16 * HDIM; i += 256) {
    int row = i >> 6, hd = i & 63;
    int q = q0 + row;
    Oh[(((size_t)b * NSEQ + q) * NHEAD + h) * HDIM + hd] = (_Float16)Obuf[i];
  }
}

// ---------------------------------------------------------------------------
// Kernel 3: output projection. out = O @ w_proj^T + b_proj (f32 out).
// M=16384, K=512, N=512. Workgroup = 64x128 tile, wave = 32x32 sub-tile.
// ---------------------------------------------------------------------------
__global__ __launch_bounds__(256) void proj_kernel(
    const _Float16* __restrict__ Oh, const _Float16* __restrict__ wh,
    const float* __restrict__ b_proj, float* __restrict__ out) {
  int lane = threadIdx.x & 31;
  int w    = threadIdx.x >> 5;
  int m0 = blockIdx.x * 64  + (w & 1) * 32;
  int n0 = blockIdx.y * 128 + (w >> 1) * 32;

  v8f acc00 = {}, acc01 = {}, acc10 = {}, acc11 = {};
  #pragma unroll 4
  for (int k0 = 0; k0 < DIM; k0 += 32) {
    v16h a0 = load_a_f16(Oh, DIM, m0,      k0, lane);
    v16h a1 = load_a_f16(Oh, DIM, m0 + 16, k0, lane);
    v16h b0 = load_b_tr_f16(wh, DIM, n0,      k0, lane);
    v16h b1 = load_b_tr_f16(wh, DIM, n0 + 16, k0, lane);
    acc00 = wmma_f16(a0, b0, acc00);
    acc01 = wmma_f16(a0, b1, acc01);
    acc10 = wmma_f16(a1, b0, acc10);
    acc11 = wmma_f16(a1, b1, acc11);
  }

  #pragma unroll
  for (int tm = 0; tm < 2; ++tm) {
    #pragma unroll
    for (int tn = 0; tn < 2; ++tn) {
      v8f acc = tm ? (tn ? acc11 : acc10) : (tn ? acc01 : acc00);
      int n = n0 + tn * 16 + (lane & 15);
      float bias = b_proj[n];
      int mbase = m0 + tm * 16 + ((lane >= 16) ? 8 : 0);
      #pragma unroll
      for (int r = 0; r < 8; ++r)
        out[(size_t)(mbase + r) * DIM + n] = acc[r] + bias;
    }
  }
}

// ---------------------------------------------------------------------------
extern "C" void kernel_launch(void* const* d_in, const int* in_sizes, int n_in,
                              void* d_out, int out_size, void* d_ws,
                              size_t ws_size, hipStream_t stream) {
  (void)in_sizes; (void)n_in; (void)out_size; (void)ws_size;
  const float*         x         = (const float*)d_in[0];
  const float*         attn_bias = (const float*)d_in[1];
  const unsigned char* mask      = (const unsigned char*)d_in[2];
  const float*         w_qkv     = (const float*)d_in[3];
  const float*         w_proj    = (const float*)d_in[4];
  const float*         b_proj    = (const float*)d_in[5];

  float* out    = (float*)d_out;                       // [16,1024,512]
  float* attn_w = out + (size_t)BATCH * NSEQ * DIM;    // [16,8,1024,1024]

  const size_t perbuf  = (size_t)BATCH * NHEAD * NSEQ * HDIM;  // 8Mi f16
  const size_t xelems  = (size_t)BATCH * NSEQ * DIM;           // 8Mi
  const size_t wqelems = (size_t)3 * DIM * DIM;                // 768Ki
  const size_t wpelems = (size_t)DIM * DIM;                    // 256Ki

  _Float16* Qh  = (_Float16*)d_ws;
  _Float16* Kh  = Qh + perbuf;
  _Float16* Vt  = Kh + perbuf;
  _Float16* Oh  = Vt + perbuf;     // [16,1024,512] f16
  _Float16* xh  = Oh + perbuf;     // [16384,512] f16
  _Float16* wqh = xh + xelems;     // [1536,512] f16
  _Float16* wph = wqh + wqelems;   // [512,512] f16

  // 0) f32 -> f16 staging of GEMM operands.
  cvt_f16_kernel<<<dim3((unsigned)(xelems / 1024)), 256, 0, stream>>>(
      x, xh, (int)xelems);
  cvt_f16_kernel<<<dim3((unsigned)(wqelems / 1024)), 256, 0, stream>>>(
      w_qkv, wqh, (int)wqelems);
  cvt_f16_kernel<<<dim3((unsigned)(wpelems / 1024)), 256, 0, stream>>>(
      w_proj, wph, (int)wpelems);

  // 1) QKV projection: grid 256 x 12 (64x128 tiles over 16384 x 1536)
  qkv_kernel<<<dim3(256, 12), 256, 0, stream>>>(xh, wqh, Qh, Kh, Vt);

  // 2) Fused attention: one block per (b, h, 16-query tile) = 8192 blocks
  size_t smem = (size_t)(16 * NSEQ + 16 * HDIM) * sizeof(float) +
                (size_t)(16 * NSEQ) * sizeof(_Float16);  // 100 KB
  attn_kernel<<<dim3(8192), 256, smem, stream>>>(attn_bias, mask, Qh, Kh, Vt,
                                                 attn_w, Oh);

  // 3) Output projection: grid 256 x 4 (64x128 tiles over 16384 x 512)
  proj_kernel<<<dim3(256, 4), 256, 0, stream>>>(Oh, wph, b_proj, out);
}
